// MoELayer_23922967838962
// MI455X (gfx1250) — compile-verified, matrix operand order checked by reading, not run on hardware
//
#include <hip/hip_runtime.h>
#include <math.h>

#define Bb 4
#define Ss 2048
#define Dd 1024
#define Ee 8
#define DFf 4096
#define NT 8192     // B*S tokens
#define CAP 8192    // per-expert token capacity (worst case)

typedef __attribute__((ext_vector_type(16))) __bf16 v16bf;
typedef __attribute__((ext_vector_type(8)))  __bf16 v8bf;
typedef __attribute__((ext_vector_type(8)))  float  v8f;

static __device__ __forceinline__ unsigned pack_bf2(float a, float b) {
#if __has_builtin(__builtin_amdgcn_cvt_pk_bf16_f32)
  auto r = __builtin_amdgcn_cvt_pk_bf16_f32(a, b);
  unsigned u;
  __builtin_memcpy(&u, &r, sizeof(unsigned));
  return u;
#else
  union { __bf16 h[2]; unsigned u; } c;
  c.h[0] = (__bf16)a; c.h[1] = (__bf16)b;
  return c.u;
#endif
}

static __device__ __forceinline__ float gelu_exact(float v) {
  return 0.5f * v * (1.0f + erff(v * 0.70710678118654752f));
}

// ---------------------------------------------------------------- utilities
__global__ void zero_small_kernel(unsigned* cnt, float* sg, float* sd) {
  int i = threadIdx.x;
  if (i < Ee) { cnt[i] = 0u; sg[i] = 0.f; sd[i] = 0.f; }
}

__global__ void zero_out_kernel(float4* p, int n4) {
  int i = blockIdx.x * blockDim.x + threadIdx.x;
  if (i < n4) p[i] = float4{0.f, 0.f, 0.f, 0.f};
}

// ---------------------------------------------------------------- router
// one wave (32 lanes) per token: 8 dot products over D=1024, softmax, top-2
__global__ __launch_bounds__(256) void router_kernel(
    const float* __restrict__ x, const float* __restrict__ Wr,
    const float* __restrict__ br, float* __restrict__ combine,
    unsigned* __restrict__ cnt, unsigned* __restrict__ list,
    float* __restrict__ Sg, float* __restrict__ Sd) {
  const int w = threadIdx.x >> 5, lane = threadIdx.x & 31;
  const int t = blockIdx.x * 8 + w;
  const float* xr = x + (size_t)t * Dd;

  float acc[Ee];
#pragma unroll
  for (int i = 0; i < Ee; ++i) acc[i] = 0.f;

  for (int d = lane; d < Dd; d += 32) {
    float xv = xr[d];
    float4 r0 = *(const float4*)(Wr + (size_t)d * Ee);
    float4 r1 = *(const float4*)(Wr + (size_t)d * Ee + 4);
    acc[0] += xv * r0.x; acc[1] += xv * r0.y; acc[2] += xv * r0.z; acc[3] += xv * r0.w;
    acc[4] += xv * r1.x; acc[5] += xv * r1.y; acc[6] += xv * r1.z; acc[7] += xv * r1.w;
  }
#pragma unroll
  for (int off = 16; off >= 1; off >>= 1) {
#pragma unroll
    for (int i = 0; i < Ee; ++i) acc[i] += __shfl_xor(acc[i], off, 32);
  }
  if (lane == 0) {
    float l[Ee], g[Ee];
    float m = -3.4e38f;
#pragma unroll
    for (int i = 0; i < Ee; ++i) { l[i] = acc[i] + br[i]; m = fmaxf(m, l[i]); }
    float s = 0.f;
#pragma unroll
    for (int i = 0; i < Ee; ++i) { g[i] = __expf(l[i] - m); s += g[i]; }
    float inv = 1.f / s;
#pragma unroll
    for (int i = 0; i < Ee; ++i) g[i] *= inv;
    // top-2 (lowest index wins ties, matching top_k)
    int i1 = 0;
#pragma unroll
    for (int i = 1; i < Ee; ++i) if (g[i] > g[i1]) i1 = i;
    int i2 = (i1 == 0) ? 1 : 0;
#pragma unroll
    for (int i = 0; i < Ee; ++i) if (i != i1 && g[i] > g[i2]) i2 = i;
#pragma unroll
    for (int i = 0; i < Ee; ++i) {
      float c = (i == i1) ? g[i1] : ((i == i2) ? g[i2] : 0.f);
      combine[(size_t)t * Ee + i] = c;
      atomicAdd(&Sg[i], g[i]);
    }
    unsigned p1 = atomicAdd(&cnt[i1], 1u); list[(size_t)i1 * CAP + p1] = (unsigned)t;
    unsigned p2 = atomicAdd(&cnt[i2], 1u); list[(size_t)i2 * CAP + p2] = (unsigned)t;
    atomicAdd(&Sd[i1], 1.f);
    atomicAdd(&Sd[i2], 1.f);
  }
}

// ================================================================ GEMMs
// Block: 256 threads (8 waves, 2x4 wave grid). Tile: 64(M) x 256(N), K-step 32.
// Per wave: 32x64 subtile -> acc[2][4], 8 v_wmma_f32_16x16x32_bf16 per K-step.

// ---------------------------------------------------------------- GEMM1
// H[row, f] = gelu( x[list[row], :] @ W1[e][:, f] + b1[e][f] ), bf16 out
__global__ __launch_bounds__(256) void gemm1_kernel(
    const float* __restrict__ x, const float* __restrict__ W1,
    const float* __restrict__ b1, const unsigned* __restrict__ cnt,
    const unsigned* __restrict__ list, __bf16* __restrict__ H, int e) {
  const int count = (int)cnt[e];
  const int mbase = blockIdx.y * 64;
  if (mbase >= count) return;
  const int nbase = blockIdx.x * 256;
  const unsigned* lst = list + (size_t)e * CAP;
  const float* W = W1 + (size_t)e * Dd * DFf;

  __shared__ __align__(16) __bf16 As[64][40];
  __shared__ __align__(16) __bf16 Bs[256][40];

  const int tid = threadIdx.x;
  const int lane = tid & 31;
  const int w = tid >> 5;
  const int wm = (w >> 2) * 32;      // 0,32
  const int wn = (w & 3) * 64;       // 0,64,128,192

  // A staging: 4 threads per row, 8 k-elements each
  const int arow = tid >> 2;
  const int akq = (tid & 3) * 8;
  int rg = mbase + arow;
  int tok = (int)lst[(rg < count) ? rg : (count - 1)];
  const float* xrow = x + (size_t)tok * Dd;
  // B staging: each thread one d-pair x 16 f-elements
  const int bd2 = tid >> 4;          // d = bd2*2   (0..15 -> d 0..30)
  const int bf0 = (tid & 15) * 16;   // f offset 0..240

  v8f acc[2][4];
#pragma unroll
  for (int i = 0; i < 2; ++i)
#pragma unroll
    for (int j = 0; j < 4; ++j) acc[i][j] = v8f{};

  for (int kb = 0; kb < Dd; kb += 32) {
    // ---- stage A (gathered x rows, fp32 -> bf16)
    {
      float4 a0 = *(const float4*)(xrow + kb + akq);
      float4 a1 = *(const float4*)(xrow + kb + akq + 4);
      unsigned* dst = (unsigned*)&As[arow][akq];
      dst[0] = pack_bf2(a0.x, a0.y);
      dst[1] = pack_bf2(a0.z, a0.w);
      dst[2] = pack_bf2(a1.x, a1.y);
      dst[3] = pack_bf2(a1.z, a1.w);
    }
    // ---- stage B transposed: W1[d][f] -> Bs[f][d], k-pairs packed
    {
      const float* w0 = W + (size_t)(kb + bd2 * 2) * DFf + nbase + bf0;
      const float* w1 = w0 + DFf;
      if (kb + 32 < Dd) __builtin_prefetch(w0 + (size_t)32 * DFf, 0, 0);
      float cf[16], df[16];
#pragma unroll
      for (int q = 0; q < 4; ++q) {
        float4 c4 = *(const float4*)(w0 + q * 4);
        float4 d4 = *(const float4*)(w1 + q * 4);
        cf[q * 4 + 0] = c4.x; cf[q * 4 + 1] = c4.y; cf[q * 4 + 2] = c4.z; cf[q * 4 + 3] = c4.w;
        df[q * 4 + 0] = d4.x; df[q * 4 + 1] = d4.y; df[q * 4 + 2] = d4.z; df[q * 4 + 3] = d4.w;
      }
#pragma unroll
      for (int i = 0; i < 16; ++i)
        *(unsigned*)&Bs[bf0 + i][bd2 * 2] = pack_bf2(cf[i], df[i]);
    }
    __syncthreads();

    // ---- fragments + WMMA
    const int fr = lane & 15;
    const int kq = (lane >> 4) * 8;   // lanes 0-15: K{0..7,16..23}; 16-31: K{8..15,24..31}
    v16bf af[2], bfr[4];
#pragma unroll
    for (int i = 0; i < 2; ++i) {
      v8bf lo = *(const v8bf*)&As[wm + i * 16 + fr][kq];
      v8bf hi = *(const v8bf*)&As[wm + i * 16 + fr][kq + 16];
      af[i] = __builtin_shufflevector(lo, hi, 0, 1, 2, 3, 4, 5, 6, 7, 8, 9, 10, 11, 12, 13, 14, 15);
    }
#pragma unroll
    for (int j = 0; j < 4; ++j) {
      v8bf lo = *(const v8bf*)&Bs[wn + j * 16 + fr][kq];
      v8bf hi = *(const v8bf*)&Bs[wn + j * 16 + fr][kq + 16];
      bfr[j] = __builtin_shufflevector(lo, hi, 0, 1, 2, 3, 4, 5, 6, 7, 8, 9, 10, 11, 12, 13, 14, 15);
    }
#pragma unroll
    for (int i = 0; i < 2; ++i)
#pragma unroll
      for (int j = 0; j < 4; ++j)
        acc[i][j] = __builtin_amdgcn_wmma_f32_16x16x32_bf16(
            false, af[i], false, bfr[j], (short)0, acc[i][j], false, false);
    __syncthreads();
  }

  // ---- epilogue: bias + exact GELU -> bf16 H
#pragma unroll
  for (int i = 0; i < 2; ++i) {
#pragma unroll
    for (int j = 0; j < 4; ++j) {
      int col = nbase + wn + j * 16 + (lane & 15);
      float bias = b1[(size_t)e * DFf + col];
      int rbase = mbase + wm + i * 16 + ((lane >> 4) * 8);
#pragma unroll
      for (int r = 0; r < 8; ++r) {
        float v = acc[i][j][r] + bias;
        H[(size_t)(rbase + r) * DFf + col] = (__bf16)gelu_exact(v);
      }
    }
  }
}

// ---------------------------------------------------------------- GEMM2
// out[tok, d] += combine[tok,e] * ( H[row,:] @ W2[e][:, d] + b2[e][d] )
__global__ __launch_bounds__(256) void gemm2_kernel(
    const __bf16* __restrict__ H, const float* __restrict__ W2,
    const float* __restrict__ b2, const float* __restrict__ combine,
    const unsigned* __restrict__ cnt, const unsigned* __restrict__ list,
    float* __restrict__ out, int e) {
  const int count = (int)cnt[e];
  const int mbase = blockIdx.y * 64;
  if (mbase >= count) return;
  const int nbase = blockIdx.x * 256;
  const unsigned* lst = list + (size_t)e * CAP;
  const float* W = W2 + (size_t)e * DFf * Dd;

  __shared__ __align__(16) __bf16 As[64][40];
  __shared__ __align__(16) __bf16 Bs[256][40];

  const int tid = threadIdx.x;
  const int lane = tid & 31;
  const int w = tid >> 5;
  const int wm = (w >> 2) * 32;
  const int wn = (w & 3) * 64;

  const int arow = tid >> 2;
  const int akq = (tid & 3) * 8;
  const int bd2 = tid >> 4;
  const int bf0 = (tid & 15) * 16;

  v8f acc[2][4];
#pragma unroll
  for (int i = 0; i < 2; ++i)
#pragma unroll
    for (int j = 0; j < 4; ++j) acc[i][j] = v8f{};

  for (int kb = 0; kb < DFf; kb += 32) {
    // ---- stage A: H rows are already bf16 and densely packed per expert
    {
      uint4 q = *(const uint4*)(H + (size_t)(mbase + arow) * DFf + kb + akq);
      *(uint4*)&As[arow][akq] = q;
    }
    // ---- stage B transposed: W2[f][d] -> Bs[d][f], k-pairs packed
    {
      const float* w0 = W + (size_t)(kb + bd2 * 2) * Dd + nbase + bf0;
      const float* w1 = w0 + Dd;
      if (kb + 32 < DFf) __builtin_prefetch(w0 + (size_t)32 * Dd, 0, 0);
      float cf[16], df[16];
#pragma unroll
      for (int q = 0; q < 4; ++q) {
        float4 c4 = *(const float4*)(w0 + q * 4);
        float4 d4 = *(const float4*)(w1 + q * 4);
        cf[q * 4 + 0] = c4.x; cf[q * 4 + 1] = c4.y; cf[q * 4 + 2] = c4.z; cf[q * 4 + 3] = c4.w;
        df[q * 4 + 0] = d4.x; df[q * 4 + 1] = d4.y; df[q * 4 + 2] = d4.z; df[q * 4 + 3] = d4.w;
      }
#pragma unroll
      for (int i = 0; i < 16; ++i)
        *(unsigned*)&Bs[bf0 + i][bd2 * 2] = pack_bf2(cf[i], df[i]);
    }
    __syncthreads();

    const int fr = lane & 15;
    const int kq = (lane >> 4) * 8;
    v16bf af[2], bfr[4];
#pragma unroll
    for (int i = 0; i < 2; ++i) {
      v8bf lo = *(const v8bf*)&As[wm + i * 16 + fr][kq];
      v8bf hi = *(const v8bf*)&As[wm + i * 16 + fr][kq + 16];
      af[i] = __builtin_shufflevector(lo, hi, 0, 1, 2, 3, 4, 5, 6, 7, 8, 9, 10, 11, 12, 13, 14, 15);
    }
#pragma unroll
    for (int j = 0; j < 4; ++j) {
      v8bf lo = *(const v8bf*)&Bs[wn + j * 16 + fr][kq];
      v8bf hi = *(const v8bf*)&Bs[wn + j * 16 + fr][kq + 16];
      bfr[j] = __builtin_shufflevector(lo, hi, 0, 1, 2, 3, 4, 5, 6, 7, 8, 9, 10, 11, 12, 13, 14, 15);
    }
#pragma unroll
    for (int i = 0; i < 2; ++i)
#pragma unroll
      for (int j = 0; j < 4; ++j)
        acc[i][j] = __builtin_amdgcn_wmma_f32_16x16x32_bf16(
            false, af[i], false, bfr[j], (short)0, acc[i][j], false, false);
    __syncthreads();
  }

  // ---- epilogue: scatter-add with combine weight (race-free: unique rows/cols)
#pragma unroll
  for (int i = 0; i < 2; ++i) {
#pragma unroll
    for (int j = 0; j < 4; ++j) {
      int col = nbase + wn + j * 16 + (lane & 15);
      float bias = b2[(size_t)e * Dd + col];
      int rbase = mbase + wm + i * 16 + ((lane >> 4) * 8);
#pragma unroll
      for (int r = 0; r < 8; ++r) {
        int rowg = rbase + r;
        if (rowg < count) {
          int tok = (int)lst[rowg];
          float c = combine[(size_t)tok * Ee + e];
          float* op = out + (size_t)tok * Dd + col;
          *op += c * (acc[i][j][r] + bias);
        }
      }
    }
  }
}

// ---------------------------------------------------------------- aux loss
__global__ void aux_kernel(const float* Sg, const float* Sd, float* out_aux) {
  if (threadIdx.x == 0) {
    double a = 0.0;
#pragma unroll
    for (int i = 0; i < Ee; ++i) a += (double)Sg[i] * (double)Sd[i];
    const double nt = (double)NT;
    out_aux[0] = (float)(a * (double)Ee / (nt * nt * nt));
  }
}

// ---------------------------------------------------------------- launch
extern "C" void kernel_launch(void* const* d_in, const int* in_sizes, int n_in,
                              void* d_out, int out_size, void* d_ws, size_t ws_size,
                              hipStream_t stream) {
  const float* x  = (const float*)d_in[0];
  const float* Wr = (const float*)d_in[1];
  const float* br = (const float*)d_in[2];
  const float* W1 = (const float*)d_in[3];
  const float* b1 = (const float*)d_in[4];
  const float* W2 = (const float*)d_in[5];
  const float* b2 = (const float*)d_in[6];
  float* out = (float*)d_out;

  char* ws = (char*)d_ws;
  unsigned* cnt     = (unsigned*)(ws + 0);
  float*    Sg      = (float*)(ws + 64);
  float*    Sd      = (float*)(ws + 128);
  float*    combine = (float*)(ws + 256);
  unsigned* list    = (unsigned*)(ws + 256 + (size_t)NT * Ee * 4);
  __bf16*   H       = (__bf16*)(ws + (1u << 20));   // CAP*DF bf16 = 64 MB

  zero_small_kernel<<<1, 64, 0, stream>>>(cnt, Sg, Sd);
  zero_out_kernel<<<(NT * Dd / 4 + 255) / 256, 256, 0, stream>>>((float4*)out, NT * Dd / 4);
  router_kernel<<<NT / 8, 256, 0, stream>>>(x, Wr, br, combine, cnt, list, Sg, Sd);
  for (int e = 0; e < Ee; ++e) {
    gemm1_kernel<<<dim3(DFf / 256, CAP / 64), 256, 0, stream>>>(x, W1, b1, cnt, list, H, e);
    gemm2_kernel<<<dim3(Dd / 256, CAP / 64), 256, 0, stream>>>(H, W2, b2, combine, cnt, list, out, e);
  }
  aux_kernel<<<1, 32, 0, stream>>>(Sg, Sd, out + (size_t)NT * Dd);
}
